// Biaffine_2_19868518712001
// MI455X (gfx1250) — matrix operand order repaired and checked by reading, not compiled, not run on hardware
//
#include <hip/hip_runtime.h>
#include <stdint.h>
#include <stddef.h>

typedef __bf16 bf16_t;
typedef __attribute__((ext_vector_type(16))) __bf16 v16bf;
typedef __attribute__((ext_vector_type(8)))  __bf16 v8bf;
typedef __attribute__((ext_vector_type(4)))  __bf16 v4bf;
typedef __attribute__((ext_vector_type(8)))  float  v8f;

#define B_ 8
#define L_ 256
#define D_ 512
#define O_ 128
#define N1_ (O_ * D_)   // 65536

#define WMMA_BF16(A, Bm, C) \
  __builtin_amdgcn_wmma_f32_16x16x32_bf16(false, (A), false, (Bm), (short)0, (C), false, false)

// ---------------- f32 -> bf16 elementwise (vectorized x4) ----------------
__global__ void cvt4_kernel(const float* __restrict__ src, bf16_t* __restrict__ dst, int n4) {
  int i = blockIdx.x * blockDim.x + threadIdx.x;
  if (i < n4) {
    float4 f = ((const float4*)src)[i];
    v4bf h;
    h[0] = (__bf16)f.x; h[1] = (__bf16)f.y; h[2] = (__bf16)f.z; h[3] = (__bf16)f.w;
    ((v4bf*)dst)[i] = h;
  }
}

// ---------------- w1[i][o][j] f32 -> w1t[o][j][i] bf16 (LDS tiled transpose) ----
__global__ void __launch_bounds__(256)
w1_transpose_kernel(const float* __restrict__ w1, bf16_t* __restrict__ w1t) {
  __shared__ float tile[32][33];
  int o  = blockIdx.x;          // 0..127
  int i0 = blockIdx.y * 32;     // i tile
  int j0 = blockIdx.z * 32;     // j tile
  int tx = threadIdx.x & 31;
  int ty = threadIdx.x >> 5;    // 0..7
  for (int ii = ty; ii < 32; ii += 8)
    tile[ii][tx] = w1[((size_t)(i0 + ii) * O_ + o) * D_ + j0 + tx];
  __syncthreads();
  for (int jj = ty; jj < 32; jj += 8)
    w1t[((size_t)o * D_ + j0 + jj) * D_ + i0 + tx] = (bf16_t)tile[tx][jj];
}

// ---------------- small projections: a2 = in1 @ w2_a (+bias), b2 = in2 @ w2_b ---
__global__ void __launch_bounds__(128)
proj_kernel(const float* __restrict__ inp, const float* __restrict__ w2part,
            const float* __restrict__ bias, float* __restrict__ outp) {
  int row = blockIdx.x;   // 0..2047 = b*L + pos
  int o   = threadIdx.x;  // 0..127
  const float* r = inp + (size_t)row * D_;
  float s = bias ? bias[o] : 0.0f;
  for (int i = 0; i < D_; ++i) s += r[i] * w2part[i * O_ + o];
  outp[(size_t)row * O_ + o] = s;
}

// ---------------- WMMA operand loaders (per documented CDNA5 VGPR layouts) -----
union ABU { v16bf v; v8bf h[2]; };

// A: 16x32 bf16.  lane<16: M=lane, elems = K{0..7},{16..23}; lane>=16: M=lane-16, K{8..15},{24..31}
__device__ __forceinline__ v16bf load_a_tile(const bf16_t* p) {
  ABU a;
  a.h[0] = *(const v8bf*)(p);
  a.h[1] = *(const v8bf*)(p + 16);
  return a.v;
}
// B: 32x16 bf16.  lane<16: N=lane, elems = K{0..7},{16..23}; lane>=16: N=lane-16, K{8..15},{24..31}
__device__ __forceinline__ v16bf load_b_tile(const bf16_t* p) {
  ABU b;
  b.h[0] = *(const v8bf*)(p);
  b.h[1] = *(const v8bf*)(p + 16);
  return b.v;
}

// ---------------- stage 1: T[m][n] = A1[m][k] * W1T[n][k]^T ---------------------
// M=2048, N=65536, K=512.  WG = 8 waves; wave tile 16(M) x 64(N), 4 acc sharing A.
// Software-pipelined: iteration i computes on tiles loaded in iteration i-1.
__global__ void __launch_bounds__(256)
stage1_gemm(const bf16_t* __restrict__ Abuf,   // in1 bf16 [2048][512]
            const bf16_t* __restrict__ Bbuf,   // w1t bf16 [65536][512]  ([n][k])
            bf16_t* __restrict__ T)            // t bf16 [2048][65536]
{
  int lane = threadIdx.x & 31;
  int wave = threadIdx.x >> 5;
  int lh = lane >> 4, ln = lane & 15;
  int m0 = blockIdx.x * 16;
  int n0 = blockIdx.y * 512 + wave * 64;

  const bf16_t* Arow = Abuf + (size_t)(m0 + ln) * D_ + lh * 8;
  const bf16_t* Bc0  = Bbuf + (size_t)(n0 + ln) * D_ + lh * 8;
  const bf16_t* Bc1  = Bc0 + (size_t)16 * D_;
  const bf16_t* Bc2  = Bc0 + (size_t)32 * D_;
  const bf16_t* Bc3  = Bc0 + (size_t)48 * D_;

  v8f acc0 = {}, acc1 = {}, acc2 = {}, acc3 = {};

  // prologue: load k=0 tiles
  v16bf a  = load_a_tile(Arow);
  v16bf b0 = load_b_tile(Bc0);
  v16bf b1 = load_b_tile(Bc1);
  v16bf b2 = load_b_tile(Bc2);
  v16bf b3 = load_b_tile(Bc3);

  #pragma unroll 3
  for (int k = 32; k < D_; k += 32) {   // 15 steady-state iterations
    v16bf an  = load_a_tile(Arow + k);
    v16bf bn0 = load_b_tile(Bc0 + k);
    v16bf bn1 = load_b_tile(Bc1 + k);
    v16bf bn2 = load_b_tile(Bc2 + k);
    v16bf bn3 = load_b_tile(Bc3 + k);
    acc0 = WMMA_BF16(a, b0, acc0);
    acc1 = WMMA_BF16(a, b1, acc1);
    acc2 = WMMA_BF16(a, b2, acc2);
    acc3 = WMMA_BF16(a, b3, acc3);
    a = an; b0 = bn0; b1 = bn1; b2 = bn2; b3 = bn3;
  }
  acc0 = WMMA_BF16(a, b0, acc0);
  acc1 = WMMA_BF16(a, b1, acc1);
  acc2 = WMMA_BF16(a, b2, acc2);
  acc3 = WMMA_BF16(a, b3, acc3);

  // D layout: VGPR r -> row m0+r (lanes 0-15) / m0+8+r (lanes 16-31), col n0+lane%16
  size_t rowBase = (size_t)(m0 + lh * 8) * N1_;
  #pragma unroll
  for (int r = 0; r < 8; ++r) {
    size_t rb = rowBase + (size_t)r * N1_ + n0 + ln;
    T[rb]      = (bf16_t)acc0[r];
    T[rb + 16] = (bf16_t)acc1[r];
    T[rb + 32] = (bf16_t)acc2[r];
    T[rb + 48] = (bf16_t)acc3[r];
  }
}

// ---------------- stage 2: out[b,x,y,o] = in2[b,y,:] . t[b,x,o,:] + a2 + b2 -----
// Per (b,x): M=256(y), N=128(o), K=512.  WG = 8 waves; wave tile 32(y) x 16(o),
// 2 acc sharing one B tile; software-pipelined like stage 1.
__global__ void __launch_bounds__(256)
stage2_gemm(const bf16_t* __restrict__ In2b,  // [2048][512]
            const bf16_t* __restrict__ T,     // [2048][128][512]  (b*L+x, o, j)
            const float* __restrict__ A2,     // [2048][128]  (bias folded in)
            const float* __restrict__ B2v,    // [2048][128]
            float* __restrict__ Out)          // [8][256][256][128]
{
  int lane = threadIdx.x & 31;
  int wave = threadIdx.x >> 5;
  int lh = lane >> 4, ln = lane & 15;
  int y0 = blockIdx.x * 32;
  int x  = blockIdx.y;
  int b  = blockIdx.z;
  int o0 = wave * 16;

  const bf16_t* Arow0 = In2b + ((size_t)(b * L_) + y0 + ln) * D_ + lh * 8;
  const bf16_t* Arow1 = Arow0 + (size_t)16 * D_;
  const bf16_t* Bc    = T + (((size_t)(b * L_ + x)) * O_ + o0 + ln) * D_ + lh * 8;

  v8f acc0 = {}, acc1 = {};

  v16bf a0 = load_a_tile(Arow0);
  v16bf a1 = load_a_tile(Arow1);
  v16bf bt = load_b_tile(Bc);

  #pragma unroll 3
  for (int k = 32; k < D_; k += 32) {   // 15 steady-state iterations
    v16bf an0 = load_a_tile(Arow0 + k);
    v16bf an1 = load_a_tile(Arow1 + k);
    v16bf bn  = load_b_tile(Bc + k);
    acc0 = WMMA_BF16(a0, bt, acc0);
    acc1 = WMMA_BF16(a1, bt, acc1);
    a0 = an0; a1 = an1; bt = bn;
  }
  acc0 = WMMA_BF16(a0, bt, acc0);
  acc1 = WMMA_BF16(a1, bt, acc1);

  int o = o0 + ln;
  float a2v = A2[(size_t)(b * L_ + x) * O_ + o];
  const float* b2p0 = B2v + (size_t)(b * L_ + y0 + lh * 8) * O_ + o;
  const float* b2p1 = b2p0 + (size_t)16 * O_;
  float* outp0 = Out + (((size_t)(b * L_ + x) * L_) + y0 + lh * 8) * O_ + o;
  float* outp1 = outp0 + (size_t)16 * O_;
  #pragma unroll
  for (int r = 0; r < 8; ++r) {
    outp0[(size_t)r * O_] = acc0[r] + a2v + b2p0[(size_t)r * O_];
    outp1[(size_t)r * O_] = acc1[r] + a2v + b2p1[(size_t)r * O_];
  }
}

extern "C" void kernel_launch(void* const* d_in, const int* in_sizes, int n_in,
                              void* d_out, int out_size, void* d_ws, size_t ws_size,
                              hipStream_t stream) {
  (void)in_sizes; (void)n_in; (void)out_size; (void)ws_size;
  const float* in1 = (const float*)d_in[0];
  const float* in2 = (const float*)d_in[1];
  const float* w1  = (const float*)d_in[2];
  const float* w2  = (const float*)d_in[3];
  float* out = (float*)d_out;

  char* ws = (char*)d_ws;
  size_t off = 0;
  bf16_t* T    = (bf16_t*)(ws + off); off += (size_t)B_ * L_ * O_ * D_ * 2;  // 256 MB
  bf16_t* W1T  = (bf16_t*)(ws + off); off += (size_t)D_ * O_ * D_ * 2;       //  64 MB
  bf16_t* IN1B = (bf16_t*)(ws + off); off += (size_t)B_ * L_ * D_ * 2;       //   2 MB
  bf16_t* IN2B = (bf16_t*)(ws + off); off += (size_t)B_ * L_ * D_ * 2;       //   2 MB
  float*  A2   = (float*)(ws + off);  off += (size_t)B_ * L_ * O_ * 4;       //   1 MB
  float*  B2v  = (float*)(ws + off);                                         //   1 MB

  const int nin = B_ * L_ * D_;  // 1048576, divisible by 4*256
  cvt4_kernel<<<nin / 4 / 256, 256, 0, stream>>>(in1, IN1B, nin / 4);
  cvt4_kernel<<<nin / 4 / 256, 256, 0, stream>>>(in2, IN2B, nin / 4);

  dim3 gT(O_, D_ / 32, D_ / 32);
  w1_transpose_kernel<<<gT, 256, 0, stream>>>(w1, W1T);

  proj_kernel<<<B_ * L_, O_, 0, stream>>>(in1, w2, w2 + 2 * D_ * O_, A2);
  proj_kernel<<<B_ * L_, O_, 0, stream>>>(in2, w2 + (size_t)D_ * O_, nullptr, B2v);

  dim3 g1(B_ * L_ / 16, N1_ / 512);
  stage1_gemm<<<g1, 256, 0, stream>>>(IN1B, W1T, T);

  dim3 g2(L_ / 32, L_, B_);
  stage2_gemm<<<g2, 256, 0, stream>>>(IN2B, T, A2, B2v, out);
}
